// BEVTrajDecoder_77386720739467
// MI455X (gfx1250) — matrix operand order repaired and checked by reading, not compiled, not run on hardware
//
#include <hip/hip_runtime.h>

// ---------------------------------------------------------------------------
// BEVTrajDecoder on gfx1250 (MI455X).
// All dense linear layers run on v_wmma_f32_16x16x32_bf16 (fp32 accumulate).
// Double-buffered LDS pipeline: one s_barrier per K-step; raw float4 loads
// stay in flight across the WMMAs (bf16 conversion deferred to store time);
// clamped row addressing (no divergent guards in the hot loop).
// T=80, K=6, NH=8, P=4, D=256, FFN_D=1024, B=8, PAST=10, HB=WB=200.
// ---------------------------------------------------------------------------

typedef __attribute__((ext_vector_type(16))) __bf16 v16bf;
typedef __attribute__((ext_vector_type(8)))  __bf16 v8bf;
typedef __attribute__((ext_vector_type(8)))  float  v8f;

#define BM 64
#define BN 64
#define BKK 32
#define LDSK 40   // row stride (elems) for LDS tiles; 40*2B=80B, keeps 16B alignment

__device__ __forceinline__ v8bf cvt8(const float4 a, const float4 b) {
    v8bf r;
    r[0] = (__bf16)a.x; r[1] = (__bf16)a.y; r[2] = (__bf16)a.z; r[3] = (__bf16)a.w;
    r[4] = (__bf16)b.x; r[5] = (__bf16)b.y; r[6] = (__bf16)b.z; r[7] = (__bf16)b.w;
    return r;
}

// ===========================================================================
// Generic GEMM: Y[M,N] = A[M,Kd] @ W[N,Kd]^T + bias  (flags: 1=relu, 2=+R)
// Row clamping is safe: OOB A rows only feed OOB C rows, OOB W rows only
// feed OOB C cols; epilogue stores are bounds-checked.
// ===========================================================================
__global__ __launch_bounds__(256)
void gemm_bias_kernel(const float* __restrict__ A, int lda,
                      const float* __restrict__ W, int ldw,
                      const float* __restrict__ bias,
                      const float* __restrict__ R,
                      float* __restrict__ Y, int ldy,
                      int M, int N, int Kd, int flags)
{
    __shared__ __bf16 As[2][BM][LDSK];
    __shared__ __bf16 Bs[2][BN][LDSK];

    const int tid  = threadIdx.x;
    const int m0   = blockIdx.x * BM;
    const int n0   = blockIdx.y * BN;
    const int wid  = tid >> 5;
    const int lane = tid & 31;
    const int wy   = wid >> 2;    // 0..1 : 32-row strip
    const int wx   = wid & 3;     // 0..3 : 16-col strip
    const int lhalf = lane >> 4;  // 0/1
    const int lm    = lane & 15;

    v8f acc0 = {}; v8f acc1 = {};

    const int lr = tid >> 2;        // 0..63 tile row
    const int lc = (tid & 3) * 8;   // 0/8/16/24 k-chunk

    const int ar = (m0 + lr < M) ? (m0 + lr) : (M - 1);   // clamped
    const int br = (n0 + lr < N) ? (n0 + lr) : (N - 1);
    const float* aptr = A + (size_t)ar * lda + lc;        // + k0 each step
    const float* wptr = W + (size_t)br * ldw + lc;

    // prologue: tile 0 -> raw registers (convert at store time)
    float4 fa0, fa1, fb0, fb1;
    { const float4* s4 = (const float4*)aptr; fa0 = s4[0]; fa1 = s4[1]; }
    { const float4* s4 = (const float4*)wptr; fb0 = s4[0]; fb1 = s4[1]; }

    int cur = 0;
    for (int k0 = 0; k0 < Kd; k0 += BKK) {
        *(v8bf*)(&As[cur][lr][lc]) = cvt8(fa0, fa1);
        *(v8bf*)(&Bs[cur][lr][lc]) = cvt8(fb0, fb1);
        __syncthreads();

        // issue next tile's global loads; data unused until next store -> no stall here
        if (k0 + BKK < Kd) {
            const float4* s4 = (const float4*)(aptr + k0 + BKK);
            fa0 = s4[0]; fa1 = s4[1];
            const float4* w4 = (const float4*)(wptr + k0 + BKK);
            fb0 = w4[0]; fb1 = w4[1];
        }

        // B fragment: lane holds col n=wx*16+lm; elems 0..7 = K lhalf*8..+7, 8..15 = K lhalf*8+16..+23
        v16bf bfr;
        {
            v8bf blo = *(const v8bf*)&Bs[cur][wx*16 + lm][lhalf*8];
            v8bf bhi = *(const v8bf*)&Bs[cur][wx*16 + lm][lhalf*8 + 16];
            #pragma unroll
            for (int i = 0; i < 8; ++i) { bfr[i] = blo[i]; bfr[8 + i] = bhi[i]; }
        }
        // A fragments (two 16x16 M-blocks); elems 0..7 = K base..+7, 8..15 = K base+16..+23
        #pragma unroll
        for (int mb = 0; mb < 2; ++mb) {
            v8bf alo = *(const v8bf*)&As[cur][wy*32 + mb*16 + lm][lhalf*8];
            v8bf ahi = *(const v8bf*)&As[cur][wy*32 + mb*16 + lm][lhalf*8 + 16];
            v16bf afr;
            #pragma unroll
            for (int i = 0; i < 8; ++i) { afr[i] = alo[i]; afr[8 + i] = ahi[i]; }
            if (mb == 0)
                acc0 = __builtin_amdgcn_wmma_f32_16x16x32_bf16(false, afr, false, bfr,
                                                               (short)0, acc0, false, false);
            else
                acc1 = __builtin_amdgcn_wmma_f32_16x16x32_bf16(false, afr, false, bfr,
                                                               (short)0, acc1, false, false);
        }
        cur ^= 1;
    }

    const int n = n0 + wx*16 + lm;
    float bv = (n < N) ? bias[n] : 0.f;
    #pragma unroll
    for (int mb = 0; mb < 2; ++mb) {
        v8f acc = mb ? acc1 : acc0;
        #pragma unroll
        for (int r = 0; r < 8; ++r) {
            int m = m0 + wy*32 + mb*16 + lhalf*8 + r;
            if (m < M && n < N) {
                float v = acc[r] + bv;
                if (flags & 1) v = fmaxf(v, 0.f);
                if (flags & 2) v += R[(size_t)m * ldy + n];
                Y[(size_t)m * ldy + n] = v;
            }
        }
    }
}

// ===========================================================================
// BEV value GEMM: A is bev[B,C,HW] (channel-major). Row m=b*HW+hw, k=c.
// Coalesced b128 loads along hw (contiguous); tiles never cross b (HW%64==0).
// Y[320000,256] = bev_hwc @ W[256,256]^T + bias.  All dims exact multiples.
// ===========================================================================
__global__ __launch_bounds__(256)
void gemm_bev_kernel(const float* __restrict__ bev,
                     const float* __restrict__ W, int ldw,
                     const float* __restrict__ bias,
                     float* __restrict__ Y,
                     int HW, int C, int N)
{
    __shared__ __bf16 As[2][BM][LDSK];
    __shared__ __bf16 Bs[2][BN][LDSK];

    const int tid  = threadIdx.x;
    const int m0   = blockIdx.x * BM;
    const int n0   = blockIdx.y * BN;
    const int b    = m0 / HW;
    const int hw0  = m0 % HW;
    const int wid  = tid >> 5;
    const int lane = tid & 31;
    const int wy   = wid >> 2;
    const int wx   = wid & 3;
    const int lhalf = lane >> 4;
    const int lm    = lane & 15;

    v8f acc0 = {}; v8f acc1 = {};

    const int ac = tid & 31;        // channel within k-tile
    const int ar = (tid >> 5) * 8;  // 8-row group along hw
    const int lr = tid >> 2;
    const int lc = (tid & 3) * 8;

    const float* bevp = bev + (size_t)b * C * HW + (size_t)ac * HW + hw0 + ar;
    const float* wptr = W + (size_t)(n0 + lr) * ldw + lc;

    // prologue: tile 0 -> raw registers
    float4 fa0, fa1, fb0, fb1;
    { const float4* s4 = (const float4*)bevp; fa0 = s4[0]; fa1 = s4[1]; }
    { const float4* s4 = (const float4*)wptr; fb0 = s4[0]; fb1 = s4[1]; }

    int cur = 0;
    for (int k0 = 0; k0 < C; k0 += BKK) {
        // transpose-store A fragment (channel-major -> [hw][c]) + vector B store
        As[cur][ar + 0][ac] = (__bf16)fa0.x; As[cur][ar + 1][ac] = (__bf16)fa0.y;
        As[cur][ar + 2][ac] = (__bf16)fa0.z; As[cur][ar + 3][ac] = (__bf16)fa0.w;
        As[cur][ar + 4][ac] = (__bf16)fa1.x; As[cur][ar + 5][ac] = (__bf16)fa1.y;
        As[cur][ar + 6][ac] = (__bf16)fa1.z; As[cur][ar + 7][ac] = (__bf16)fa1.w;
        *(v8bf*)(&Bs[cur][lr][lc]) = cvt8(fb0, fb1);
        __syncthreads();

        // stream the next tile (unconditional speculative prefetch one tile further)
        __builtin_prefetch(bevp + (size_t)(k0 + 2 * BKK) * HW, 0, 0);
        if (k0 + BKK < C) {
            const float4* s4 = (const float4*)(bevp + (size_t)(k0 + BKK) * HW);
            fa0 = s4[0]; fa1 = s4[1];
            const float4* w4 = (const float4*)(wptr + k0 + BKK);
            fb0 = w4[0]; fb1 = w4[1];
        }

        v16bf bfr;
        {
            v8bf blo = *(const v8bf*)&Bs[cur][wx*16 + lm][lhalf*8];
            v8bf bhi = *(const v8bf*)&Bs[cur][wx*16 + lm][lhalf*8 + 16];
            #pragma unroll
            for (int i = 0; i < 8; ++i) { bfr[i] = blo[i]; bfr[8 + i] = bhi[i]; }
        }
        #pragma unroll
        for (int mb = 0; mb < 2; ++mb) {
            v8bf alo = *(const v8bf*)&As[cur][wy*32 + mb*16 + lm][lhalf*8];
            v8bf ahi = *(const v8bf*)&As[cur][wy*32 + mb*16 + lm][lhalf*8 + 16];
            v16bf afr;
            #pragma unroll
            for (int i = 0; i < 8; ++i) { afr[i] = alo[i]; afr[8 + i] = ahi[i]; }
            if (mb == 0)
                acc0 = __builtin_amdgcn_wmma_f32_16x16x32_bf16(false, afr, false, bfr,
                                                               (short)0, acc0, false, false);
            else
                acc1 = __builtin_amdgcn_wmma_f32_16x16x32_bf16(false, afr, false, bfr,
                                                               (short)0, acc1, false, false);
        }
        cur ^= 1;
    }

    const int n = n0 + wx*16 + lm;
    float bv = bias[n];
    #pragma unroll
    for (int mb = 0; mb < 2; ++mb) {
        v8f acc = mb ? acc1 : acc0;
        #pragma unroll
        for (int r = 0; r < 8; ++r) {
            int m = m0 + wy*32 + mb*16 + lhalf*8 + r;
            Y[(size_t)m * N + n] = acc[r] + bv;
        }
    }
}

// ===========================================================================
// Flash-style MHA: one wave per (sq, nb, h); lane = head-dim channel (dh=32).
// Token layout seq-first: row = s*NB + nb. O row width 256, col h*32+d.
// ===========================================================================
__global__ __launch_bounds__(32)
void attn_kernel(const float* __restrict__ Q, int q_ld, int q_off,
                 const float* __restrict__ Kt, int k_ld, int k_off,
                 const float* __restrict__ V,  int v_ld, int v_off,
                 float* __restrict__ O,
                 int Sq, int Sk, int NB, float scale)
{
    int idx = blockIdx.x;
    int h  = idx & 7;  idx >>= 3;
    int nb = idx % NB;
    int sq = idx / NB;
    int d  = threadIdx.x;

    float qv = Q[((size_t)sq * NB + nb) * q_ld + q_off + h*32 + d] * scale;

    float mmax = -1e30f, l = 0.f, acc = 0.f;
    for (int t = 0; t < Sk; ++t) {
        float p = qv * Kt[((size_t)t * NB + nb) * k_ld + k_off + h*32 + d];
        #pragma unroll
        for (int o = 16; o > 0; o >>= 1) p += __shfl_xor(p, o, 32);
        float nm   = fmaxf(mmax, p);
        float corr = __expf(mmax - nm);
        float e    = __expf(p - nm);
        float vv   = V[((size_t)t * NB + nb) * v_ld + v_off + h*32 + d];
        l   = l * corr + e;
        acc = acc * corr + e * vv;
        mmax = nm;
    }
    O[((size_t)sq * NB + nb) * 256 + h*32 + d] = acc / l;
}

// ===========================================================================
// LayerNorm over D=256 (one row per block). Optional pre-add residual,
// optional fused [b,k,t] -> [k,b,t] output permute (for the final store).
// ===========================================================================
__global__ __launch_bounds__(256)
void ln_kernel(const float* __restrict__ X, const float* __restrict__ Res,
               const float* __restrict__ g, const float* __restrict__ bta,
               float* __restrict__ Y, int permute)
{
    __shared__ float red[256];
    __shared__ float stat[2];
    int row = blockIdx.x, d = threadIdx.x;
    float x = X[(size_t)row * 256 + d];
    if (Res) x += Res[(size_t)row * 256 + d];

    red[d] = x; __syncthreads();
    for (int s = 128; s > 0; s >>= 1) { if (d < s) red[d] += red[d + s]; __syncthreads(); }
    if (d == 0) stat[0] = red[0] * (1.f / 256.f);
    __syncthreads();
    float c = x - stat[0];
    red[d] = c * c; __syncthreads();
    for (int s = 128; s > 0; s >>= 1) { if (d < s) red[d] += red[d + s]; __syncthreads(); }
    if (d == 0) stat[1] = red[0] * (1.f / 256.f);
    __syncthreads();
    float y = c * __frsqrt_rn(stat[1] + 1e-5f) * g[d] + bta[d];

    int orow = row;
    if (permute) { // row = b*480 + k*80 + t  ->  k*640 + b*80 + t
        int b = row / 480, rem = row % 480, k = rem / 80, t = rem % 80;
        orow = k * 640 + b * 80 + t;
    }
    Y[(size_t)orow * 256 + d] = y;
}

// ===========================================================================
// Small elementwise / gather kernels
// ===========================================================================
__global__ void add_kernel(const float* __restrict__ a, const float* __restrict__ b,
                           float* __restrict__ c, int n)
{
    int i = blockIdx.x * blockDim.x + threadIdx.x;
    if (i < n) c[i] = a[i] + b[i];
}

// DETR-style 2D sine embedding. ref [K,B,T,2] flat row = k*640+b*80+t.
__global__ __launch_bounds__(256)
void sine_kernel(const float* __restrict__ ref, float* __restrict__ out)
{
    int row = blockIdx.x, d = threadIdx.x;
    float p = (d < 128) ? ref[row * 2 + 1] : ref[row * 2 + 0]; // emb(y) || emb(x)
    int j = d & 127;
    int m = j >> 1;
    float dim_t = __powf(10000.f, (2.f * (float)m) * (1.f / 128.f));
    float e = p * 6.28318530717958647692f / dim_t;
    out[(size_t)row * 256 + d] = (j & 1) ? __cosf(e) : __sinf(e);
}

// [T, B*K, D] rows -> [K,B,T] rows
__global__ __launch_bounds__(256)
void reorder_tbk_kernel(const float* __restrict__ X, float* __restrict__ Y)
{
    int row = blockIdx.x, d = threadIdx.x;  // out row (k,b,t)
    int k = row / 640, rem = row % 640, b = rem / 80, t = rem % 80;
    Y[(size_t)row * 256 + d] = X[(size_t)((t * 8 + b) * 6 + k) * 256 + d];
}

// ref_points [K,B,T,2] -> ego frame, out rows b*480 + k*80 + t
__global__ void ego_kernel(const float* __restrict__ ref,
                           const float* __restrict__ ex, const float* __restrict__ ey,
                           const float* __restrict__ es, const float* __restrict__ ec,
                           float* __restrict__ out)
{
    int i = blockIdx.x * blockDim.x + threadIdx.x;
    if (i >= 3840) return;
    int b = i / 480, rem = i % 480, k = rem / 80, t = rem % 80;
    size_t ri = (size_t)((k * 8 + b) * 80 + t) * 2;
    float x = ref[ri], y = ref[ri + 1];
    float s = es[b], c = ec[b];
    out[(size_t)i * 2 + 0] = x * c - y * s + ex[b];
    out[(size_t)i * 2 + 1] = x * s + y * c + ey[b];
}

// qmod = x * query_scale; also emit xflat. Out rows b*480+k*80+t.
__global__ __launch_bounds__(256)
void qmod_kernel(const float* __restrict__ Xkbt, const float* __restrict__ qs_tbk,
                 float* __restrict__ qmod, float* __restrict__ xflat)
{
    int row = blockIdx.x, d = threadIdx.x;
    int b = row / 480, rem = row % 480, k = rem / 80, t = rem % 80;
    float xv = Xkbt[(size_t)(k * 640 + b * 80 + t) * 256 + d];
    float q  = qs_tbk[(size_t)(t * 48 + b * 6 + k) * 256 + d];
    xflat[(size_t)row * 256 + d] = xv;
    qmod [(size_t)row * 256 + d] = xv * q;
}

// Deformable bilinear sampling. One wave per (b,q,h); lane = dh channel.
__global__ __launch_bounds__(32)
void dca_sample_kernel(const float* __restrict__ Vv,   // [B*40000, 256]
                       const float* __restrict__ off,  // [3840, 64]
                       const float* __restrict__ awr,  // [3840, 32] (pre-softmax)
                       const float* __restrict__ refE, // [3840, 2]
                       float* __restrict__ O)          // [3840, 256]
{
    int idx = blockIdx.x;
    int h = idx & 7, row = idx >> 3;        // row = b*480 + q
    int b = row / 480;
    int d = threadIdx.x;

    float a0 = awr[(size_t)row * 32 + h * 4 + 0];
    float a1 = awr[(size_t)row * 32 + h * 4 + 1];
    float a2 = awr[(size_t)row * 32 + h * 4 + 2];
    float a3 = awr[(size_t)row * 32 + h * 4 + 3];
    float mx = fmaxf(fmaxf(a0, a1), fmaxf(a2, a3));
    float e0 = __expf(a0 - mx), e1 = __expf(a1 - mx);
    float e2 = __expf(a2 - mx), e3 = __expf(a3 - mx);
    float inv = 1.f / (e0 + e1 + e2 + e3);
    float aw[4] = { e0 * inv, e1 * inv, e2 * inv, e3 * inv };

    float rx = refE[(size_t)row * 2 + 0];
    float ry = refE[(size_t)row * 2 + 1];
    const float* vb = Vv + (size_t)b * 40000 * 256 + h * 32 + d;

    float acc = 0.f;
    #pragma unroll
    for (int p = 0; p < 4; ++p) {
        float ox = off[(size_t)row * 64 + (h * 4 + p) * 2 + 0];
        float oy = off[(size_t)row * 64 + (h * 4 + p) * 2 + 1];
        float xp = (rx + ox * (1.f / 200.f)) * 200.f - 0.5f;
        float yp = (ry + oy * (1.f / 200.f)) * 200.f - 0.5f;
        float x0f = floorf(xp), y0f = floorf(yp);
        float fx = xp - x0f, fy = yp - y0f;
        int x0 = (int)x0f, y0 = (int)y0f;
        float s = 0.f;
        #pragma unroll
        for (int cy = 0; cy < 2; ++cy)
            #pragma unroll
            for (int cx = 0; cx < 2; ++cx) {
                int xi = x0 + cx, yi = y0 + cy;
                float w = (cx ? fx : 1.f - fx) * (cy ? fy : 1.f - fy);
                bool valid = (xi >= 0) && (xi < 200) && (yi >= 0) && (yi < 200);
                int xc = xi < 0 ? 0 : (xi > 199 ? 199 : xi);
                int yc = yi < 0 ? 0 : (yi > 199 ? 199 : yi);
                float gv = vb[(size_t)(yc * 200 + xc) * 256];
                s += valid ? (w * gv) : 0.f;
            }
        acc += aw[p] * s;
    }
    O[(size_t)row * 256 + h * 32 + d] = acc;
}

// ===========================================================================
// Host orchestration
// ===========================================================================
extern "C" void kernel_launch(void* const* d_in, const int* in_sizes, int n_in,
                              void* d_out, int out_size, void* d_ws, size_t ws_size,
                              hipStream_t stream)
{
    const float* dec   = (const float*)d_in[0];   // [80, 48, 256]
    const float* scene = (const float*)d_in[1];   // [10, 640, 256]
    const float* bev   = (const float*)d_in[2];   // [8, 256, 200, 200]
    const float* qsc   = (const float*)d_in[3];   // [80, 48, 256]
    const float* refp  = (const float*)d_in[4];   // [6, 8, 80, 2]
    const float* ex    = (const float*)d_in[5];
    const float* ey    = (const float*)d_in[6];
    const float* esn   = (const float*)d_in[7];
    const float* ecs   = (const float*)d_in[8];
    const float* tpe   = (const float*)d_in[9];
    const float* tsa_in_w  = (const float*)d_in[10]; const float* tsa_in_b  = (const float*)d_in[11];
    const float* tsa_out_w = (const float*)d_in[12]; const float* tsa_out_b = (const float*)d_in[13];
    const float* posq_w1 = (const float*)d_in[14]; const float* posq_b1 = (const float*)d_in[15];
    const float* posq_w2 = (const float*)d_in[16]; const float* posq_b2 = (const float*)d_in[17];
    const float* ln0_g = (const float*)d_in[18]; const float* ln0_b = (const float*)d_in[19];
    const float* sa_in_w  = (const float*)d_in[20]; const float* sa_in_b  = (const float*)d_in[21];
    const float* sa_out_w = (const float*)d_in[22]; const float* sa_out_b = (const float*)d_in[23];
    const float* ca_in_w  = (const float*)d_in[24]; const float* ca_in_b  = (const float*)d_in[25];
    const float* ca_out_w = (const float*)d_in[26]; const float* ca_out_b = (const float*)d_in[27];
    const float* tff_w1 = (const float*)d_in[28]; const float* tff_b1 = (const float*)d_in[29];
    const float* tff_w2 = (const float*)d_in[30]; const float* tff_b2 = (const float*)d_in[31];
    const float* tln1_g = (const float*)d_in[32]; const float* tln1_b = (const float*)d_in[33];
    const float* tln2_g = (const float*)d_in[34]; const float* tln2_b = (const float*)d_in[35];
    const float* tln3_g = (const float*)d_in[36]; const float* tln3_b = (const float*)d_in[37];
    const float* off_w = (const float*)d_in[38]; const float* off_b = (const float*)d_in[39];
    const float* aw_w  = (const float*)d_in[40]; const float* aw_b  = (const float*)d_in[41];
    const float* val_w = (const float*)d_in[42]; const float* val_b = (const float*)d_in[43];
    const float* dout_w = (const float*)d_in[44]; const float* dout_b = (const float*)d_in[45];
    const float* ln1_g = (const float*)d_in[46]; const float* ln1_b = (const float*)d_in[47];
    const float* ffn_w1 = (const float*)d_in[48]; const float* ffn_b1 = (const float*)d_in[49];
    const float* ffn_w2 = (const float*)d_in[50]; const float* ffn_b2 = (const float*)d_in[51];
    const float* ln2_g = (const float*)d_in[52]; const float* ln2_b = (const float*)d_in[53];

    float* out = (float*)d_out;

    // workspace layout (float units)
    const size_t NV = 81920000ull;  // 320000*256 value buffer
    const size_t NHB = 3932160ull;  // 3840*1024 (also holds 3840*768 / 6400*512)
    const size_t NS = 983040ull;    // 3840*256
    if (ws_size < (NV + NHB + 8 * NS) * sizeof(float)) return;

    float* V  = (float*)d_ws;
    float* H  = V + NV;
    float* S0 = H + NHB;
    float* S1 = S0 + NS; float* S2 = S1 + NS; float* S3 = S2 + NS;
    float* S4 = S3 + NS; float* S5 = S4 + NS; float* S6 = S5 + NS; float* S7 = S6 + NS;

    const float scale = 0.17677669529663689f; // 1/sqrt(32)
    const int M = 3840;                       // T*B*K = K*B*T

    auto gemm = [&](const float* A, int lda, const float* W, int ldw, const float* bias,
                    const float* R, float* Y, int ldy, int Mm, int Nn, int Kd, int flags) {
        dim3 g((Mm + BM - 1) / BM, (Nn + BN - 1) / BN);
        gemm_bias_kernel<<<g, dim3(256), 0, stream>>>(A, lda, W, ldw, bias, R, Y, ldy, Mm, Nn, Kd, flags);
    };

    // ---- Stage 1: temporal self-attention (S=80, NB=48) ----
    add_kernel<<<(M * 256 + 255) / 256, 256, 0, stream>>>(dec, tpe, S0, M * 256);
    gemm(S0, 256, tsa_in_w, 256, tsa_in_b, nullptr, H, 512, M, 512, 256, 0);            // q|k
    gemm(dec, 256, tsa_in_w + 512 * 256, 256, tsa_in_b + 512, nullptr, S1, 256, M, 256, 256, 0); // v
    attn_kernel<<<80 * 48 * 8, 32, 0, stream>>>(H, 512, 0, H, 512, 256, S1, 256, 0, S2, 80, 80, 48, scale);
    gemm(S2, 256, tsa_out_w, 256, tsa_out_b, dec, S3, 256, M, 256, 256, 2);             // + residual
    ln_kernel<<<M, 256, 0, stream>>>(S3, nullptr, ln0_g, ln0_b, S4, 0);                 // x [t,b*k]

    // ---- Stage 2: positional query ----
    sine_kernel<<<M, 256, 0, stream>>>(refp, S0);                                       // [k,b,t] rows
    gemm(S0, 256, posq_w1, 256, posq_b1, nullptr, S1, 256, M, 256, 256, 1);             // relu
    reorder_tbk_kernel<<<M, 256, 0, stream>>>(S4, S2);                                  // x -> [k,b,t]
    gemm(S1, 256, posq_w2, 256, posq_b2, S2, S3, 256, M, 256, 256, 2);                  // tgt = x + pos_q

    // ---- Stage 3: transformer decoder layer (seq=6, NB=640) ----
    gemm(S3, 256, sa_in_w, 256, sa_in_b, nullptr, H, 768, M, 768, 256, 0);              // qkv
    attn_kernel<<<6 * 640 * 8, 32, 0, stream>>>(H, 768, 0, H, 768, 256, H, 768, 512, S0, 6, 6, 640, scale);
    gemm(S0, 256, sa_out_w, 256, sa_out_b, S3, S1, 256, M, 256, 256, 2);
    ln_kernel<<<M, 256, 0, stream>>>(S1, nullptr, tln1_g, tln1_b, S2, 0);
    gemm(S2, 256, ca_in_w, 256, ca_in_b, nullptr, S0, 256, M, 256, 256, 0);             // q
    gemm(scene, 256, ca_in_w + 256 * 256, 256, ca_in_b + 256, nullptr, H, 512, 6400, 512, 256, 0); // k|v
    attn_kernel<<<6 * 640 * 8, 32, 0, stream>>>(S0, 256, 0, H, 512, 0, H, 512, 256, S1, 6, 10, 640, scale);
    gemm(S1, 256, ca_out_w, 256, ca_out_b, S2, S3, 256, M, 256, 256, 2);
    ln_kernel<<<M, 256, 0, stream>>>(S3, nullptr, tln2_g, tln2_b, S0, 0);
    gemm(S0, 256, tff_w1, 256, tff_b1, nullptr, H, 1024, M, 1024, 256, 1);              // relu
    gemm(H, 1024, tff_w2, 1024, tff_b2, S0, S1, 256, M, 256, 1024, 2);
    ln_kernel<<<M, 256, 0, stream>>>(S1, nullptr, tln3_g, tln3_b, S2, 0);               // x [k,b,t]

    // ---- Stage 4: ego transform + query-scale modulation ----
    ego_kernel<<<15, 256, 0, stream>>>(refp, ex, ey, esn, ecs, S3);                     // [b, k*t, 2]
    qmod_kernel<<<M, 256, 0, stream>>>(S2, qsc, S4, S5);                                // qmod, xflat

    // ---- Stage 5: BEV deformable cross-attention ----
    gemm_bev_kernel<<<dim3(5000, 4), 256, 0, stream>>>(bev, val_w, 256, val_b, V, 40000, 256, 256);
    gemm(S4, 256, off_w, 256, off_b, nullptr, S0, 64, M, 64, 256, 0);                   // offsets
    gemm(S4, 256, aw_w, 256, aw_b, nullptr, S1, 32, M, 32, 256, 0);                     // attn weights
    dca_sample_kernel<<<M * 8, 32, 0, stream>>>(V, S0, S1, S3, S6);
    gemm(S6, 256, dout_w, 256, dout_b, S5, S7, 256, M, 256, 256, 2);                    // + xflat
    ln_kernel<<<M, 256, 0, stream>>>(S7, nullptr, ln1_g, ln1_b, S0, 0);

    // ---- Stage 6: final FFN + LN (fused [b,kt]->[k,b,t] permute) ----
    gemm(S0, 256, ffn_w1, 256, ffn_b1, nullptr, H, 1024, M, 1024, 256, 1);
    gemm(H, 1024, ffn_w2, 1024, ffn_b2, S0, S1, 256, M, 256, 1024, 2);
    ln_kernel<<<M, 256, 0, stream>>>(S1, nullptr, ln2_g, ln2_b, out, 1);
}
